// SoftArmGraphVLA_70506183131142
// MI455X (gfx1250) — matrix operand order repaired and checked by reading, not compiled, not run on hardware
//
#include <hip/hip_runtime.h>
#include <hip/hip_bf16.h>

typedef _Float16 f16;
typedef __attribute__((ext_vector_type(16))) _Float16 v16h;
typedef __attribute__((ext_vector_type(8)))  _Float16 v8h;
typedef __attribute__((ext_vector_type(8)))  float    v8f;
typedef unsigned int u32x4 __attribute__((ext_vector_type(4)));
typedef int          i32x4 __attribute__((ext_vector_type(4)));
typedef int          i32x8 __attribute__((ext_vector_type(8)));

#define BATCH    128
#define CONV_K   147      // 3*7*7 true K
#define KPAD     168      // c*56 + kh*8 + kw  (kw padded 7->8)
#define KSTEPS   6        // ceil(168/32) -> k-space 0..191, weights 0-padded past 168
#define FC1_IN   1568     // 49 k-steps of 32
#define FC1_OUT  512

#if defined(__HIP_DEVICE_COMPILE__) && __has_builtin(__builtin_amdgcn_tensor_load_to_lds)
#define HAVE_TDM 1
#else
#define HAVE_TDM 0
#endif

// ---------------------------------------------------------------- utilities
__global__ void zero_kernel(float* p, int n) {
    int t = blockIdx.x * blockDim.x + threadIdx.x;
    if (t < n) p[t] = 0.0f;
}

__global__ void convert_f16_kernel(const float* __restrict__ src, f16* __restrict__ dst, int n) {
    int t = blockIdx.x * blockDim.x + threadIdx.x;
    if (t < n) dst[t] = (f16)src[t];
}

// Pre-swizzle conv weights into a contiguous-k (k = ks*32 + 16*(lane/16) + j) 16-bit A-matrix (16x32) fragment
// layout over the padded K-space kpad = c*56 + kh*8 + kw (0-fill kw==7 and
// kpad >= 168), so each lane loads its 16 halves with one contiguous 32B read.
// wfrag index = ((ks*2 + mtile)*32 + lane)*16 + j
__global__ void rearrange_conv_w_kernel(const float* __restrict__ w, f16* __restrict__ wfrag) {
    int t = blockIdx.x * blockDim.x + threadIdx.x;   // 6*2*32*16 = 6144
    if (t >= KSTEPS * 2 * 32 * 16) return;
    int j    = t & 15;
    int lane = (t >> 4) & 31;
    int mt   = (t >> 9) & 1;
    int ks   = t >> 10;
    int lh   = lane >> 4;
    int m    = mt * 16 + (lane & 15);
    int kpad = ks * 32 + lh * 16 + j;
    float v = 0.0f;
    if (kpad < KPAD) {
        int c = kpad / 56, rem = kpad % 56;
        int kh = rem >> 3, kw = rem & 7;
        if (kw < 7) v = w[m * CONV_K + c * 49 + kh * 7 + kw];
    }
    wfrag[t] = (f16)v;
}

// ----------------------------------------------------- conv + relu + 16x16 pool
// One block per (batch, pool_row, pool_col).  Implicit GEMM via WMMA:
//   D[ch][pos] = sum_k W[ch][k] * im2col[k][pos]   (pos over the 16x16 pool cell)
// Patch staged by the Tensor Data Mover, im2col expanded once into LDS, bias
// folded into the accumulator, relu + average-pool fused at the end.
__global__ __launch_bounds__(256)
void conv_pool_kernel(const float* __restrict__ img, const f16* __restrict__ wfrag,
                      const float* __restrict__ conv_b,
                      float* __restrict__ pooled_f32, f16* __restrict__ pooled_f16) {
    int blk   = blockIdx.x;           // b*49 + pr*7 + pc
    int b     = blk / 49;
    int pcell = blk % 49;
    int pr = pcell / 7, pc = pcell % 7;
    int ir0 = pr * 32 - 3;            // input row of local conv output (0,0)
    int ic0 = pc * 32 - 3;

    // software clip of the 37x37 patch against the 224x224 image (conv pad=3)
    int rskip = ir0 < 0 ? -ir0 : 0;
    int cskip = ic0 < 0 ? -ic0 : 0;
    int rend  = (224 - ir0) < 37 ? (224 - ir0) : 37;
    int cend  = (224 - ic0) < 37 ? (224 - ic0) : 37;
    int W = cend - cskip, H = rend - rskip;

    __shared__ __attribute__((aligned(16))) float patchf[3 * 37 * 37];   // packed W*H per channel
    __shared__ __attribute__((aligned(32))) f16   imcol[128 * KPAD + 64]; // [pos][kpad] + read-tail pad
    __shared__ float acc[32];

    int tid = threadIdx.x;
    if (tid < 32) acc[tid] = 0.0f;

#if HAVE_TDM
    if (tid == 0) {
        unsigned lds_base = (unsigned)(uintptr_t)&patchf[0];   // flat LDS addr low 32 = LDS offset
        const float* gp = img + ((long long)(b * 3) * 224 + (ir0 + rskip)) * 224 + (ic0 + cskip);
        unsigned long long ga = (unsigned long long)(uintptr_t)gp;
        u32x4 g0;
        g0[0] = 1u;                                            // count=1, user descriptor
        g0[1] = lds_base;
        g0[2] = (unsigned)(ga & 0xFFFFFFFFu);
        g0[3] = ((unsigned)(ga >> 32) & 0x01FFFFFFu) | 0x80000000u;  // type=2 ("image")
        i32x8 g1;
        g1[0] = (int)(2u << 16);                               // data_size = 4 bytes
        g1[1] = (int)((unsigned)(W & 0xFFFF) << 16);           // tensor_dim0[15:0]
        g1[2] = (int)((unsigned)(H & 0xFFFF) << 16);           // td0[31:16]=0 | tensor_dim1[15:0]
        g1[3] = (int)((unsigned)(W & 0xFFFF) << 16);           // td1[31:16]=0 | tile_dim0
        g1[4] = (int)((unsigned)(H & 0xFFFF) | (3u << 16));    // tile_dim1 | tile_dim2=3
        g1[5] = 224;                                           // tensor_dim0_stride low32
        g1[6] = (int)(((unsigned)(224 * 224) & 0xFFFFu) << 16);// td0s hi16=0 | td1_stride[15:0]
        g1[7] = (int)((unsigned)(224 * 224) >> 16);            // td1_stride[47:16]
        i32x4 g2; g2[0] = 3; g2[1] = 0; g2[2] = 0; g2[3] = 0;  // tensor_dim2=3, no dim3
        i32x4 g3; g3[0] = 0; g3[1] = 0; g3[2] = 0; g3[3] = 0;
#if defined(__clang_major__) && (__clang_major__ >= 23)
        i32x8 g4; for (int i = 0; i < 8; ++i) g4[i] = 0;
        __builtin_amdgcn_tensor_load_to_lds(g0, g1, g2, g3, g4, 0);
#else
        __builtin_amdgcn_tensor_load_to_lds(g0, g1, g2, g3, 0);
#endif
    }
    __builtin_amdgcn_s_wait_tensorcnt(0);   // no-op for waves that issued nothing
    __syncthreads();
#else
    { // manual staging fallback, identical packed layout
        int total = 3 * H * W;
        for (int t = tid; t < total; t += 256) {
            int c  = t / (H * W);
            int rr = (t / W) % H;
            int cc = t % W;
            patchf[t] = img[((long long)(b * 3 + c) * 224 + (ir0 + rskip + rr)) * 224 + (ic0 + cskip + cc)];
        }
        __syncthreads();
    }
#endif

    int lane  = tid & 31;
    int wid   = tid >> 5;
    int lhalf = lane >> 4;
    int nlo   = lane & 15;
    int chbase = lhalf * 8;
    int lp = wid * 16 + nlo;          // this wave's local position within a half

    for (int hlf = 0; hlf < 2; ++hlf) {
        // ---- expand im2col for positions [hlf*128, hlf*128+128) into LDS ----
        for (int t = tid; t < 128 * 21; t += 256) {
            int p  = t / 21;          // local pos 0..127
            int ck = t % 21;
            int c = ck / 7, kh = ck % 7;
            int gpos = hlf * 128 + p;
            int r = gpos >> 4, col = gpos & 15;
            int row  = 2 * r + kh;                 // patch-local row 0..36
            int prow = row - rskip;
            bool rok = (prow >= 0) && (row < rend);
            const float* prp = &patchf[(c * H + prow) * W] - cskip;
            v8h pk;
            for (int kw = 0; kw < 7; ++kw) {       // contiguous 7-f32 run in the patch row
                int cc = 2 * col + kw;
                float v = 0.0f;
                if (rok && cc >= cskip && cc < cend) v = prp[cc];
                pk[kw] = (f16)v;
            }
            pk[7] = (f16)0.0f;
            *(v8h*)&imcol[p * KPAD + c * 56 + kh * 8] = pk;   // 16B-aligned store
        }
        __syncthreads();

        // ---- per-wave WMMA GEMM over this half's 8 N-tiles (one per wave) ----
        v8f c0, c1;
        for (int i = 0; i < 8; ++i) {              // bias into accumulator (pre-relu)
            c0[i] = conv_b[chbase + i];
            c1[i] = conv_b[16 + chbase + i];
        }
        for (int ks = 0; ks < KSTEPS; ++ks) {
            v16h a0 = *(const v16h*)(wfrag + ((ks * 2 + 0) * 32 + lane) * 16);
            v16h a1 = *(const v16h*)(wfrag + ((ks * 2 + 1) * 32 + lane) * 16);
            const v8h* bp = (const v8h*)&imcol[lp * KPAD + ks * 32 + lhalf * 16];
            v8h b0 = bp[0];                         // ds_load_b128 x2, zero gather math
            v8h b1 = bp[1];
            v16h bf;
            for (int j = 0; j < 8; ++j) { bf[j] = b0[j]; bf[8 + j] = b1[j]; }
            c0 = __builtin_amdgcn_wmma_f32_16x16x32_f16(false, a0, false, bf, (short)0, c0, false, false);
            c1 = __builtin_amdgcn_wmma_f32_16x16x32_f16(false, a1, false, bf, (short)0, c1, false, false);
        }
        // relu then accumulate into the pool sum (ds_add_f32)
        for (int i = 0; i < 8; ++i) {
            atomicAdd(&acc[chbase + i],      fmaxf(c0[i], 0.0f));
            atomicAdd(&acc[16 + chbase + i], fmaxf(c1[i], 0.0f));
        }
        __syncthreads();   // protect imcol before the next half rebuilds it
    }

    if (tid < 32) {
        float v = acc[tid] * (1.0f / 256.0f);
        int o = b * FC1_IN + tid * 49 + pcell;   // (B, 32*49): ch*49 + pr*7 + pc
        pooled_f32[o] = v;
        pooled_f16[o] = (f16)v;
    }
}

// ----------------------------------------------------------- img_fc1 via WMMA
// out[b][o] = relu( sum_k pooled[b][k] * w[o][k] + bias[o] ),  M=128 N=512 K=1568
__global__ __launch_bounds__(256)
void fc1_wmma_kernel(const f16* __restrict__ x16, const f16* __restrict__ w16,
                     const float* __restrict__ bias, float* __restrict__ out) {
    int lane  = threadIdx.x & 31;
    int wid   = threadIdx.x >> 5;
    int mtile = blockIdx.x;              // 0..7
    int ntile = blockIdx.y * 8 + wid;    // 0..31
    int lhalf = lane >> 4;
    int nlo   = lane & 15;
    int o = ntile * 16 + nlo;
    int m = mtile * 16 + nlo;

    v8f c;
    float bv = bias[o];
    for (int i = 0; i < 8; ++i) c[i] = bv;

    for (int ks = 0; ks < FC1_IN / 32; ++ks) {
        // A fragment: gather batch-row activations, contiguous k = 16*(lane/16) + j as in B
        v16h a;
        for (int j = 0; j < 16; ++j) {
            int k = ks * 32 + lhalf * 16 + j;
            a[j] = x16[m * FC1_IN + k];
        }
        // B fragment: weights are K-contiguous per output -> two 16B vector loads
        const v8h* bp = (const v8h*)(w16 + o * FC1_IN + ks * 32 + lhalf * 16);
        v8h b0 = bp[0];
        v8h b1 = bp[1];
        v16h bf;
        for (int j = 0; j < 8; ++j) { bf[j] = b0[j]; bf[8 + j] = b1[j]; }
        c = __builtin_amdgcn_wmma_f32_16x16x32_f16(false, a, false, bf, (short)0, c, false, false);
    }

    int mrow = mtile * 16 + lhalf * 8;
    for (int i = 0; i < 8; ++i)
        out[(mrow + i) * FC1_OUT + o] = fmaxf(c[i], 0.0f);
}

// ---------------------------------------------- vlm = fc2(fc1out) + text_fc(emb)
__global__ void vlm_kernel(const float* __restrict__ fc1out,
                           const float* __restrict__ fc2w, const float* __restrict__ fc2b,
                           const int* __restrict__ tlen, const float* __restrict__ temb,
                           const float* __restrict__ tw, const float* __restrict__ tb,
                           float* __restrict__ vlm) {
    int t = blockIdx.x * blockDim.x + threadIdx.x;
    if (t >= BATCH * 768) return;
    int b = t / 768, o = t % 768;
    float s = fc2b[o] + tb[o];
    const float* xr = fc1out + b * 512;
    const float* wr = fc2w + o * 512;
    for (int k = 0; k < 512; ++k) s += xr[k] * wr[k];
    int ti = tlen[b];
    ti = ti < 0 ? 0 : (ti > 9999 ? 9999 : ti);
    const float* er  = temb + ti * 256;
    const float* twr = tw + o * 256;
    for (int k = 0; k < 256; ++k) s += er[k] * twr[k];
    vlm[t] = s;
}

// --------------------------------------------------------------- graph encoder
__global__ __launch_bounds__(128)
void graph_kernel(const float* __restrict__ nfeat, const int* __restrict__ eidx,
                  const float* __restrict__ e1w, const float* __restrict__ e1b,
                  const float* __restrict__ e2w, const float* __restrict__ e2b,
                  const float* __restrict__ gcw, const float* __restrict__ gcb,
                  const float* __restrict__ lns, const float* __restrict__ lnb,
                  const float* __restrict__ pw,  const float* __restrict__ pb,
                  const float* __restrict__ a1w, const float* __restrict__ a1b,
                  const float* __restrict__ a2w, const float* __restrict__ a2b,
                  float* __restrict__ tokens_g, float* __restrict__ robot_g) {
    int b = blockIdx.x;
    int tid = threadIdx.x;
    __shared__ float nf[10 * 19];
    __shared__ float x[10 * 64], tmp[10 * 64], mnb[10 * 64], yb[10 * 64];
    __shared__ float tok[10 * 32];
    __shared__ int   srcs[18], dsts[18];
    __shared__ float cnt[10];    // also reused as per-node mean in LN
    __shared__ float attw[10];   // also reused as per-node inv-std in LN

    for (int t = tid; t < 190; t += 128) nf[t] = nfeat[b * 190 + t];
    if (tid < 18) {
        srcs[tid] = eidx[b * 36 + tid];
        dsts[tid] = eidx[b * 36 + 18 + tid];
    }
    __syncthreads();
    // enc1 (19 -> 64, relu)
    for (int t = tid; t < 640; t += 128) {
        int n = t >> 6, o = t & 63;
        float s = e1b[o];
        for (int i = 0; i < 19; ++i) s += nf[n * 19 + i] * e1w[o * 19 + i];
        tmp[t] = fmaxf(s, 0.0f);
    }
    __syncthreads();
    // enc2 (64 -> 64)
    for (int t = tid; t < 640; t += 128) {
        int n = t >> 6, o = t & 63;
        float s = e2b[o];
        for (int i = 0; i < 64; ++i) s += tmp[n * 64 + i] * e2w[o * 64 + i];
        x[t] = s;
    }
    __syncthreads();

    for (int layer = 0; layer < 3; ++layer) {
        if (tid < 10) {
            float c = 0.0f;
            for (int e = 0; e < 18; ++e) if (dsts[e] == tid) c += 1.0f;
            cnt[tid] = fmaxf(c, 1.0f);
        }
        __syncthreads();
        // mean neighbor aggregation (segment_sum over dst)
        for (int t = tid; t < 640; t += 128) {
            int n = t >> 6, f = t & 63;
            float s = 0.0f;
            for (int e = 0; e < 18; ++e) if (dsts[e] == n) s += x[srcs[e] * 64 + f];
            mnb[t] = s / cnt[n];
        }
        __syncthreads();
        const float* wl = gcw + layer * 64 * 128;
        const float* bl = gcb + layer * 64;
        for (int t = tid; t < 640; t += 128) {
            int n = t >> 6, o = t & 63;
            float s = bl[o];
            for (int i = 0; i < 64; ++i) s += x[n * 64 + i]   * wl[o * 128 + i];
            for (int i = 0; i < 64; ++i) s += mnb[n * 64 + i] * wl[o * 128 + 64 + i];
            yb[t] = s + x[t];    // + residual
        }
        __syncthreads();
        if (tid < 10) {          // layernorm stats per node
            float mu = 0.0f;
            for (int i = 0; i < 64; ++i) mu += yb[tid * 64 + i];
            mu *= (1.0f / 64.0f);
            float var = 0.0f;
            for (int i = 0; i < 64; ++i) { float d = yb[tid * 64 + i] - mu; var += d * d; }
            var *= (1.0f / 64.0f);
            cnt[tid]  = mu;
            attw[tid] = rsqrtf(var + 1e-5f);
        }
        __syncthreads();
        const float* sl  = lns + layer * 64;
        const float* bbl = lnb + layer * 64;
        for (int t = tid; t < 640; t += 128) {
            int n = t >> 6, o = t & 63;
            float v = (yb[t] - cnt[n]) * attw[n] * sl[o] + bbl[o];
            x[t] = fmaxf(v, 0.0f);
        }
        __syncthreads();
    }

    // projection to tokens (64 -> 32)
    for (int t = tid; t < 320; t += 128) {
        int n = t >> 5, o = t & 31;
        float s = pb[o];
        for (int i = 0; i < 64; ++i) s += x[n * 64 + i] * pw[o * 64 + i];
        tok[t] = s;
    }
    __syncthreads();
    // attention scores s[n] = attn2(tanh(attn1(tok[n])))
    if (tid < 10) {
        float s = a2b[0];
        for (int u = 0; u < 16; ++u) {
            float h = a1b[u];
            for (int i = 0; i < 32; ++i) h += tok[tid * 32 + i] * a1w[u * 32 + i];
            s += tanhf(h) * a2w[u];
        }
        attw[tid] = s;
    }
    __syncthreads();
    if (tid == 0) {               // softmax over the 10 nodes
        float mx = attw[0];
        for (int n = 1; n < 10; ++n) mx = fmaxf(mx, attw[n]);
        float den = 0.0f;
        for (int n = 0; n < 10; ++n) { attw[n] = __expf(attw[n] - mx); den += attw[n]; }
        float inv = 1.0f / den;
        for (int n = 0; n < 10; ++n) attw[n] *= inv;
    }
    __syncthreads();
    if (tid < 32) {
        float s = 0.0f;
        for (int n = 0; n < 10; ++n) s += tok[n * 32 + tid] * attw[n];
        robot_g[b * 32 + tid] = s;
    }
    for (int t = tid; t < 320; t += 128) tokens_g[b * 320 + t] = tok[t];
}

// ----------------------------------------------------------------- joint heads
__global__ __launch_bounds__(128)
void head_kernel(const float* __restrict__ vlm, const float* __restrict__ robot_g,
                 const float* __restrict__ tokens_g,
                 const float* __restrict__ w1, const float* __restrict__ b1,
                 const float* __restrict__ w2, const float* __restrict__ b2,
                 const float* __restrict__ w3, const float* __restrict__ b3,
                 float* __restrict__ out) {
    int b = blockIdx.x, j = blockIdx.y;
    int tid = threadIdx.x;
    __shared__ float comb[800];
    __shared__ float h1[128], h2[64];
    for (int t = tid; t < 768; t += 128) {
        float v = vlm[b * 768 + t];
        if (t < 32) v += robot_g[b * 32 + t];     // fused = vlm[:, :32] += robot
        comb[t] = v;
    }
    if (tid < 32) comb[768 + tid] = tokens_g[b * 320 + j * 32 + tid];
    __syncthreads();
    {
        float s = b1[j * 128 + tid];
        const float* wr = w1 + (j * 128 + tid) * 800;
        for (int i = 0; i < 800; ++i) s += comb[i] * wr[i];
        h1[tid] = fmaxf(s, 0.0f);
    }
    __syncthreads();
    if (tid < 64) {
        float s = b2[j * 64 + tid];
        const float* wr = w2 + (j * 64 + tid) * 128;
        for (int i = 0; i < 128; ++i) s += h1[i] * wr[i];
        h2[tid] = fmaxf(s, 0.0f);
    }
    __syncthreads();
    if (tid < 16) {
        float s = b3[j * 16 + tid];
        const float* wr = w3 + (j * 16 + tid) * 64;
        for (int i = 0; i < 64; ++i) s += h2[i] * wr[i];
        out[b * 160 + tid * 10 + j] = s;   // (B, 16, 10) after transpose; j>=6 stays 0
    }
}

// ------------------------------------------------------------------- launcher
extern "C" void kernel_launch(void* const* d_in, const int* in_sizes, int n_in,
                              void* d_out, int out_size, void* d_ws, size_t ws_size,
                              hipStream_t stream) {
    const float* images = (const float*)d_in[0];
    const int*   tlen   = (const int*)  d_in[1];
    const float* nfeat  = (const float*)d_in[2];
    const int*   eidx   = (const int*)  d_in[3];
    const float* conv_w = (const float*)d_in[4];
    const float* conv_b = (const float*)d_in[5];
    const float* fc1w   = (const float*)d_in[6];
    const float* fc1b   = (const float*)d_in[7];
    const float* fc2w   = (const float*)d_in[8];
    const float* fc2b   = (const float*)d_in[9];
    const float* temb   = (const float*)d_in[10];
    const float* tw     = (const float*)d_in[11];
    const float* tb     = (const float*)d_in[12];
    const float* e1w    = (const float*)d_in[13];
    const float* e1b    = (const float*)d_in[14];
    const float* e2w    = (const float*)d_in[15];
    const float* e2b    = (const float*)d_in[16];
    const float* gcw    = (const float*)d_in[17];
    const float* gcb    = (const float*)d_in[18];
    const float* lns    = (const float*)d_in[19];
    const float* lnb    = (const float*)d_in[20];
    const float* pw     = (const float*)d_in[21];
    const float* pb     = (const float*)d_in[22];
    const float* a1w    = (const float*)d_in[23];
    const float* a1b    = (const float*)d_in[24];
    const float* a2w    = (const float*)d_in[25];
    const float* a2b    = (const float*)d_in[26];
    const float* w1     = (const float*)d_in[27];
    const float* b1     = (const float*)d_in[28];
    const float* w2     = (const float*)d_in[29];
    const float* b2     = (const float*)d_in[30];
    const float* w3     = (const float*)d_in[31];
    const float* b3     = (const float*)d_in[32];

    char* ws = (char*)d_ws;
    f16*   wfrag    = (f16*)  (ws + 0);        //   12,288 B
    f16*   fc1w16   = (f16*)  (ws + 16384);    // 1,605,632 B
    float* pooled   = (float*)(ws + 1622016);  //   802,816 B
    f16*   pooled16 = (f16*)  (ws + 2424832);  //   401,408 B
    float* fc1out   = (float*)(ws + 2826240);  //   262,144 B
    float* vlmb     = (float*)(ws + 3088384);  //   393,216 B
    float* tokg     = (float*)(ws + 3481600);  //   163,840 B
    float* robg     = (float*)(ws + 3645440);  //    16,384 B
    float* out = (float*)d_out;

    zero_kernel<<<(20480 + 255) / 256, 256, 0, stream>>>(out, 20480);
    rearrange_conv_w_kernel<<<(KSTEPS * 2 * 32 * 16 + 255) / 256, 256, 0, stream>>>(conv_w, wfrag);
    convert_f16_kernel<<<(512 * 1568 + 255) / 256, 256, 0, stream>>>(fc1w, fc1w16, 512 * 1568);
    conv_pool_kernel<<<BATCH * 49, 256, 0, stream>>>(images, wfrag, conv_b, pooled, pooled16);
    fc1_wmma_kernel<<<dim3(8, 4), 256, 0, stream>>>(pooled16, fc1w16, fc1b, fc1out);
    vlm_kernel<<<(BATCH * 768 + 255) / 256, 256, 0, stream>>>(fc1out, fc2w, fc2b, tlen, temb, tw, tb, vlmb);
    graph_kernel<<<BATCH, 128, 0, stream>>>(nfeat, eidx, e1w, e1b, e2w, e2b, gcw, gcb,
                                            lns, lnb, pw, pb, a1w, a1b, a2w, a2b, tokg, robg);
    head_kernel<<<dim3(BATCH, 6), 128, 0, stream>>>(vlmb, robg, tokg, w1, b1, w2, b2, w3, b3, out);
}
